// FASTKAGIN_6640019439795
// MI455X (gfx1250) — compile-verified, matrix operand order checked by reading, not run on hardware
//
#include <hip/hip_runtime.h>

// FastKAN-GIN fused pipeline for gfx1250 (MI455X).
//  - Spline + base GEMMs via v_wmma_f32_16x16x32_bf16 (fp32 accumulate)
//  - RBF basis computed on the fly into LDS (never materialized in HBM)
//  - Edge scatter-add / graph pooling via global float atomics (L2-resident table)

#define DIN   128
#define DOUT  128
#define NG    8
#define KSPL  (DIN * NG)        // 1024
#define MB    64                // rows per workgroup in the WMMA kernel
#define EPSN  1e-5f
#define RDEN  (7.0f / 4.0f)     // 1/DENOM, DENOM = 4/(G-1) = 4/7

typedef __attribute__((ext_vector_type(8)))  float  v8f;
typedef __attribute__((ext_vector_type(16))) __bf16 v16bf;
typedef __attribute__((ext_vector_type(8)))  __bf16 bf16x8;

union FragBF { v16bf v; bf16x8 h[2]; };

__device__ __forceinline__ float wred32(float v) {
#pragma unroll
  for (int m = 16; m >= 1; m >>= 1) v += __shfl_xor(v, m, 32);
  return v;
}

// ---------------------------------------------------------------- utilities
__global__ void k_cvt_bf16(const float* __restrict__ src, __bf16* __restrict__ dst, int n) {
  int i = blockIdx.x * 256 + threadIdx.x;
  if (i < n) dst[i] = (__bf16)src[i];
}

__global__ void k_fill0(float* __restrict__ p, int n) {
  int i = blockIdx.x * 256 + threadIdx.x;
  if (i < n) p[i] = 0.0f;
}

// GIN aggregation: agg[dst] += h[src], 4 floats per thread (float4 gather)
__global__ void k_scatter(const float* __restrict__ h, const int* __restrict__ srcI,
                          const int* __restrict__ dstI, float* __restrict__ agg, int nE) {
  int idx = blockIdx.x * 256 + threadIdx.x;
  int e = idx >> 5;
  if (e >= nE) return;
  int c = (idx & 31) << 2;
  int s = srcI[e], d = dstI[e];
  const float4 v = *(const float4*)(h + (size_t)s * DIN + c);
  float* p = agg + (size_t)d * DIN + c;
  atomicAdd(p + 0, v.x); atomicAdd(p + 1, v.y);
  atomicAdd(p + 2, v.z); atomicAdd(p + 3, v.w);
}

// ------------------------------------------------- fused FastKAN sublayer
// out[M][128] = basis(LN(x+agg)) @ spline_w^T + silu(x+agg) @ base_w^T + b
// 256 threads = 8 waves; 64 rows per block; wave w owns output cols [16w,16w+16)
__global__ __launch_bounds__(256)
void k_fastkan_wmma(const float* __restrict__ x, const float* __restrict__ agg,
                    const float* __restrict__ ln_g, const float* __restrict__ ln_b,
                    const __bf16* __restrict__ wspl,   // [DOUT][KSPL] bf16 row-major
                    const __bf16* __restrict__ wbase,  // [DOUT][DIN]  bf16 row-major
                    const float* __restrict__ bb, float* __restrict__ out) {
  __shared__ __attribute__((aligned(16))) float  s_xn[MB][DIN];     // 32 KB
  __shared__ __attribute__((aligned(16))) __bf16 s_silu[MB][DIN];   // 16 KB
  __shared__ __attribute__((aligned(16))) __bf16 s_basis[MB][64];   //  8 KB

  const int tid = threadIdx.x, lane = tid & 31, wave = tid >> 5;
  const int m0 = blockIdx.x * MB;
  const int kh = lane >> 4;        // which K-half group this lane holds
  const int kh8 = kh * 8;
  const int nl = lane & 15;        // row (A) / col (B) index within tile

  // ---- Stage 1: z = x (+ agg); LayerNorm -> s_xn; SiLU(z) -> s_silu
#pragma unroll
  for (int rl = 0; rl < MB / 8; ++rl) {
    const int row = wave * (MB / 8) + rl;
    const size_t base = (size_t)(m0 + row) * DIN;
    float v[4]; float s = 0.f, s2 = 0.f;
#pragma unroll
    for (int j = 0; j < 4; ++j) {
      const int c = j * 32 + lane;
      float z = x[base + c];
      if (agg) z += agg[base + c];
      v[j] = z; s += z; s2 += z * z;
    }
    s = wred32(s); s2 = wred32(s2);
    const float mu = s * (1.0f / DIN);
    const float rs = rsqrtf(s2 * (1.0f / DIN) - mu * mu + EPSN);
#pragma unroll
    for (int j = 0; j < 4; ++j) {
      const int c = j * 32 + lane;
      const float z = v[j];
      s_xn[row][c]   = (z - mu) * rs * ln_g[c] + ln_b[c];
      s_silu[row][c] = (__bf16)(z / (1.0f + __expf(-z)));
    }
  }
  __syncthreads();

  v8f acc[4] = {};   // 4 M-subtiles of 16x16 fp32

  // ---- Stage 2: spline GEMM, K = 1024 in 16 chunks of 64 (2 WMMA K-steps)
  const __bf16* wrow = wspl + (size_t)(wave * 16 + nl) * KSPL;
  const int brow  = tid >> 2;   // basis row this thread fills (0..63)
  const int bpart = tid & 3;    // 16-value slice within the 64-wide chunk
  for (int kc = 0; kc < 16; ++kc) {
    // recompute Gaussian RBF basis chunk for d in [kc*8, kc*8+8)
    const float xn0 = s_xn[brow][kc * 8 + bpart * 2 + 0];
    const float xn1 = s_xn[brow][kc * 8 + bpart * 2 + 1];
#pragma unroll
    for (int q = 0; q < NG; ++q) {
      const float gc = -2.0f + (float)q * (4.0f / 7.0f);
      const float t0 = (xn0 - gc) * RDEN;
      const float t1 = (xn1 - gc) * RDEN;
      s_basis[brow][bpart * 16 + q]     = (__bf16)__expf(-t0 * t0);
      s_basis[brow][bpart * 16 + 8 + q] = (__bf16)__expf(-t1 * t1);
    }
    __syncthreads();
#pragma unroll
    for (int h2 = 0; h2 < 2; ++h2) {
      FragBF b;   // weight fragment: lane nl = output col, two 16B loads
      const __bf16* wp = wrow + kc * 64 + h2 * 32;
      b.h[0] = *(const bf16x8*)(wp + kh8);
      b.h[1] = *(const bf16x8*)(wp + 16 + kh8);
#pragma unroll
      for (int sN = 0; sN < 4; ++sN) {
        FragBF a;  // basis fragment from LDS (ISA 16-bit A 16x32 layout)
        const __bf16* sp = &s_basis[sN * 16 + nl][h2 * 32];
        a.h[0] = *(const bf16x8*)(sp + kh8);
        a.h[1] = *(const bf16x8*)(sp + 16 + kh8);
        acc[sN] = __builtin_amdgcn_wmma_f32_16x16x32_bf16(
            false, a.v, false, b.v, (short)0, acc[sN], false, false);
      }
    }
    __syncthreads();
  }

  // ---- Stage 3: base branch GEMM, K = 128 in 4 K-steps
  const __bf16* brw = wbase + (size_t)(wave * 16 + nl) * DIN;
#pragma unroll
  for (int kc = 0; kc < 4; ++kc) {
    FragBF b;
    const __bf16* wp = brw + kc * 32;
    b.h[0] = *(const bf16x8*)(wp + kh8);
    b.h[1] = *(const bf16x8*)(wp + 16 + kh8);
#pragma unroll
    for (int sN = 0; sN < 4; ++sN) {
      FragBF a;
      const __bf16* sp = &s_silu[sN * 16 + nl][kc * 32];
      a.h[0] = *(const bf16x8*)(sp + kh8);
      a.h[1] = *(const bf16x8*)(sp + 16 + kh8);
      acc[sN] = __builtin_amdgcn_wmma_f32_16x16x32_bf16(
          false, a.v, false, b.v, (short)0, acc[sN], false, false);
    }
  }

  // ---- Epilogue: C/D layout — lane holds (M = v + 8*kh, N = nl)
  const int n = wave * 16 + nl;
  const float bbias = bb[n];
#pragma unroll
  for (int sN = 0; sN < 4; ++sN) {
#pragma unroll
    for (int vv = 0; vv < 8; ++vv) {
      const int gr = m0 + sN * 16 + kh * 8 + vv;
      out[(size_t)gr * DOUT + n] = acc[sN][vv] + bbias;
    }
  }
}

// -------------------------------------------------------------- BatchNorm
__global__ void k_bn_stats(const float* __restrict__ h, const float* __restrict__ g,
                           const float* __restrict__ b, float* __restrict__ st, int M) {
  __shared__ float sh[2][256];
  const int f = blockIdx.x, tid = threadIdx.x;
  float s = 0.f, s2 = 0.f;
  for (int r = tid; r < M; r += 256) {
    const float v = h[(size_t)r * DIN + f];
    s += v; s2 += v * v;
  }
  sh[0][tid] = s; sh[1][tid] = s2;
  __syncthreads();
  for (int stp = 128; stp >= 1; stp >>= 1) {
    if (tid < stp) { sh[0][tid] += sh[0][tid + stp]; sh[1][tid] += sh[1][tid + stp]; }
    __syncthreads();
  }
  if (tid == 0) {
    const float mu  = sh[0][0] / (float)M;
    const float var = sh[1][0] / (float)M - mu * mu;
    const float sc  = g[f] * rsqrtf(var + EPSN);
    st[f] = sc; st[DIN + f] = b[f] - mu * sc;
  }
}

__global__ void k_bn_apply(float* __restrict__ h, const float* __restrict__ st, int n) {
  int i = blockIdx.x * 256 + threadIdx.x;
  if (i >= n) return;
  const int f = i & (DIN - 1);
  h[i] = h[i] * st[f] + st[DIN + f];
}

// ------------------------------------------------------------ graph pool
__global__ void k_pool(const float* __restrict__ h, const int* __restrict__ batch,
                       float* __restrict__ pooled, int n) {
  int i = blockIdx.x * 256 + threadIdx.x;
  if (i >= n) return;
  const int r = i >> 7, f = i & 127;
  atomicAdd(&pooled[(size_t)batch[r] * DIN + f], h[i]);
}

// ------------------------- final KAN sublayer (128 -> 10) + log_softmax
__global__ void k_final(const float* __restrict__ pin, const float* __restrict__ ln_g,
                        const float* __restrict__ ln_b, const float* __restrict__ wspl,
                        const float* __restrict__ wbase, const float* __restrict__ bb,
                        float* __restrict__ out) {
  __shared__ float s_basis[KSPL];
  __shared__ float s_silu[DIN];
  const int g = blockIdx.x, lane = threadIdx.x;
  float v[4]; float s = 0.f, s2 = 0.f;
#pragma unroll
  for (int j = 0; j < 4; ++j) {
    const int c = j * 32 + lane;
    const float z = pin[(size_t)g * DIN + c];
    v[j] = z; s += z; s2 += z * z;
  }
  s = wred32(s); s2 = wred32(s2);
  const float mu = s * (1.0f / DIN);
  const float rs = rsqrtf(s2 * (1.0f / DIN) - mu * mu + EPSN);
#pragma unroll
  for (int j = 0; j < 4; ++j) {
    const int c = j * 32 + lane;
    const float z = v[j];
    const float xn = (z - mu) * rs * ln_g[c] + ln_b[c];
    s_silu[c] = z / (1.0f + __expf(-z));
#pragma unroll
    for (int q = 0; q < NG; ++q) {
      const float gc = -2.0f + (float)q * (4.0f / 7.0f);
      const float t = (xn - gc) * RDEN;
      s_basis[c * NG + q] = __expf(-t * t);
    }
  }
  __syncthreads();
  float logits[10];
#pragma unroll
  for (int o = 0; o < 10; ++o) {
    float a = 0.f;
    for (int k = lane; k < KSPL; k += 32) a += s_basis[k] * wspl[o * KSPL + k];
    for (int k = lane; k < DIN;  k += 32) a += s_silu[k]  * wbase[o * DIN + k];
    a = wred32(a);
    logits[o] = a + bb[o];
  }
  float mx = logits[0];
#pragma unroll
  for (int o = 1; o < 10; ++o) mx = fmaxf(mx, logits[o]);
  float se = 0.f;
#pragma unroll
  for (int o = 0; o < 10; ++o) se += __expf(logits[o] - mx);
  const float lse = mx + __logf(se);
  if (lane == 0) {
#pragma unroll
    for (int o = 0; o < 10; ++o) out[g * 10 + o] = logits[o] - lse;
  }
}

// ---------------------------------------------------------------- driver
extern "C" void kernel_launch(void* const* d_in, const int* in_sizes, int n_in,
                              void* d_out, int out_size, void* d_ws, size_t ws_size,
                              hipStream_t stream) {
  (void)n_in; (void)out_size; (void)ws_size;

  // Resolve flattening order: insertion-order dicts vs pytree-sorted keys.
  const bool topSorted = (in_sizes[0] < 1000000);  // x is 5.12M elems, batch is 40K
  int xi, eii, bati, convBase, bnBase, kanBase;
  bool nestedSorted;
  if (!topSorted) {
    xi = 0; eii = 1; bati = 2; convBase = 3; bnBase = 33; kanBase = 39;
    nestedSorted = (in_sizes[convBase + 1] != DIN);  // base_w(16384) if sorted, ln_b(128) if not
  } else {
    bati = 0; bnBase = 1; convBase = 7; eii = 37; kanBase = 38; xi = 48;
    nestedSorted = true;
  }
  int o_lng, o_lnb, o_spl, o_bw, o_bb, o_g, o_b;
  if (nestedSorted) { o_bb = 0; o_bw = 1; o_lnb = 2; o_lng = 3; o_spl = 4; o_b = 0; o_g = 1; }
  else              { o_lng = 0; o_lnb = 1; o_spl = 2; o_bw = 3; o_bb = 4; o_g = 0; o_b = 1; }

  const float* x     = (const float*)d_in[xi];
  const int*   ei    = (const int*)d_in[eii];
  const int*   batch = (const int*)d_in[bati];
  const int nE = in_sizes[eii] / 2;
  const int* srcI = ei;
  const int* dstI = ei + nE;
  const int M = in_sizes[xi] / DIN;          // 40000 (multiple of 64)

  int subBase[8];
  for (int i = 0; i < 6; ++i) subBase[i] = convBase + i * 5;
  subBase[6] = kanBase; subBase[7] = kanBase + 5;

  // Workspace carve (256B aligned)
  char* ws = (char*)d_ws;
  size_t off = 0;
  auto carve = [&](size_t bytes) -> void* {
    void* p = ws + off;
    off += (bytes + 255) & ~(size_t)255;
    return p;
  };
  float* agg     = (float*)carve((size_t)M * DIN * 4);
  float* bufA    = (float*)carve((size_t)M * DIN * 4);
  float* bufB    = (float*)carve((size_t)M * DIN * 4);
  float* pooled  = (float*)carve((size_t)128 * DIN * 4);
  float* pooled2 = (float*)carve((size_t)128 * DIN * 4);
  float* st      = (float*)carve(2 * DIN * 4);
  __bf16* wspl[7]; __bf16* wbas[7];
  for (int k = 0; k < 7; ++k) {
    wspl[k] = (__bf16*)carve((size_t)DOUT * KSPL * 2);
    wbas[k] = (__bf16*)carve((size_t)DOUT * DIN * 2);
  }

  // Convert WMMA-path weights (6 conv sublayers + kan sublayer 0) to bf16
  for (int k = 0; k < 7; ++k) {
    const float* ps = (const float*)d_in[subBase[k] + o_spl];
    const float* pb = (const float*)d_in[subBase[k] + o_bw];
    k_cvt_bf16<<<(DOUT * KSPL + 255) / 256, 256, 0, stream>>>(ps, wspl[k], DOUT * KSPL);
    k_cvt_bf16<<<(DOUT * DIN  + 255) / 256, 256, 0, stream>>>(pb, wbas[k], DOUT * DIN);
  }

  const int nElem = M * DIN;
  const float* h = x;
  for (int L = 0; L < 3; ++L) {
    k_fill0<<<(nElem + 255) / 256, 256, 0, stream>>>(agg, nElem);
    k_scatter<<<(unsigned)(((size_t)nE * 32 + 255) / 256), 256, 0, stream>>>(h, srcI, dstI, agg, nE);
    const int s0 = subBase[L * 2], s1 = subBase[L * 2 + 1];
    k_fastkan_wmma<<<M / MB, 256, 0, stream>>>(
        h, agg,
        (const float*)d_in[s0 + o_lng], (const float*)d_in[s0 + o_lnb],
        wspl[L * 2], wbas[L * 2], (const float*)d_in[s0 + o_bb], bufA);
    k_fastkan_wmma<<<M / MB, 256, 0, stream>>>(
        bufA, nullptr,
        (const float*)d_in[s1 + o_lng], (const float*)d_in[s1 + o_lnb],
        wspl[L * 2 + 1], wbas[L * 2 + 1], (const float*)d_in[s1 + o_bb], bufB);
    k_bn_stats<<<DIN, 256, 0, stream>>>(
        bufB, (const float*)d_in[bnBase + L * 2 + o_g],
        (const float*)d_in[bnBase + L * 2 + o_b], st, M);
    k_bn_apply<<<(nElem + 255) / 256, 256, 0, stream>>>(bufB, st, nElem);
    h = bufB;
  }

  k_fill0<<<(128 * DIN + 255) / 256, 256, 0, stream>>>(pooled, 128 * DIN);
  k_pool<<<(nElem + 255) / 256, 256, 0, stream>>>(h, batch, pooled, nElem);

  const int k0 = subBase[6], k1 = subBase[7];
  k_fastkan_wmma<<<128 / MB, 256, 0, stream>>>(
      pooled, nullptr,
      (const float*)d_in[k0 + o_lng], (const float*)d_in[k0 + o_lnb],
      wspl[6], wbas[6], (const float*)d_in[k0 + o_bb], pooled2);
  k_final<<<128, 32, 0, stream>>>(
      pooled2, (const float*)d_in[k1 + o_lng], (const float*)d_in[k1 + o_lnb],
      (const float*)d_in[k1 + o_spl], (const float*)d_in[k1 + o_bw],
      (const float*)d_in[k1 + o_bb], (float*)d_out);
}
